// EmbModule_62775241998338
// MI455X (gfx1250) — matrix operand order repaired and checked by reading, not compiled, not run on hardware
//
#include <hip/hip_runtime.h>
#include <hip/hip_bf16.h>
#include <math.h>

#define N_NODES 50000
#define NUME    10000
#define R       9
#define NH      4
#define HD      32
#define DIM_IN  128
#define DIM_T   32
#define DIN     160      // DIM_IN + DIM_T
#define DOUT    128      // NH*HD
#define NEDGE   100000
#define NROOT   4096
#define NEG_SLOPE 0.2f

typedef __attribute__((ext_vector_type(16))) __bf16 v16bf;
typedef __attribute__((ext_vector_type(8)))  __bf16 v8bf;
typedef __attribute__((ext_vector_type(8)))  float  v8f;

// ---- monotone float <-> uint encoding for atomicMax-based segment_max ----
__device__ __forceinline__ unsigned fenc(float f) {
    unsigned u = __float_as_uint(f);
    return (u & 0x80000000u) ? ~u : (u | 0x80000000u);
}
__device__ __forceinline__ float fdec(unsigned u) {
    return (u & 0x80000000u) ? __uint_as_float(u ^ 0x80000000u)
                             : __uint_as_float(~u);
}
// fenc(-inf) == 0x007FFFFFu  (init value for the max buffer)

// ---------------------------------------------------------------- kernels --

__global__ void k_convert_w(const float* __restrict__ Wl,
                            __bf16* __restrict__ Wbf, int n) {
    int i = blockIdx.x * blockDim.x + threadIdx.x;
    if (i < n) Wbf[i] = (__bf16)Wl[i];
}

__global__ void k_zero_f32(float* __restrict__ p, int n) {
    int i = blockIdx.x * blockDim.x + threadIdx.x;
    if (i < n) p[i] = 0.0f;
}

// h_cat (bf16, N x 160): cols [0,128) = emb_table[nid % NUME], cols [128,160) = time enc
__global__ void k_build_hcat(const float* __restrict__ emb,
                             const int*   __restrict__ nid,
                             const float* __restrict__ freq,
                             const float* __restrict__ phase,
                             const int*   __restrict__ tsp,
                             __bf16*      __restrict__ hcat) {
    int i = blockIdx.x * blockDim.x + threadIdx.x;
    if (i >= N_NODES * DIN) return;
    int n = i / DIN, c = i % DIN;
    int id = nid[n];
    float v;
    if (c < DIM_IN) {
        v = emb[(size_t)(id % NUME) * DIM_IN + c];
    } else {
        int j = c - DIM_IN;
        float t = (float)(tsp[0] - id / NUME);
        v = cosf(t * freq[j] + phase[j]);
    }
    hcat[i] = (__bf16)v;
}

__global__ void k_init_ms(unsigned* __restrict__ menc, float* __restrict__ sbuf) {
    int i = blockIdx.x * blockDim.x + threadIdx.x;
    if (i >= N_NODES * NH) return;
    menc[i] = 0x007FFFFFu;   // fenc(-inf)
    sbuf[i] = 0.0f;
}

// z = hcat(bf16, Nx160) @ Wr(bf16, 160x128), f32 accumulate via WMMA.
// 4 waves / block, each wave computes one 16x16 tile with 5 chained
// v_wmma_f32_16x16x32_bf16 ops (K = 160 = 5*32).
__global__ __launch_bounds__(128) void k_gemm_z(const __bf16* __restrict__ hcat,
                                                const __bf16* __restrict__ Wr,
                                                float* __restrict__ z) {
    int wid  = threadIdx.x >> 5;
    int lane = threadIdx.x & 31;
    int tile = blockIdx.x * 4 + wid;        // 0 .. 24999
    int tn = tile & 7;                      // col tile (128/16 = 8)
    int tm = tile >> 3;                     // row tile (50000/16 = 3125)
    int half = lane >> 4;                   // 0 / 1
    int m = tm * 16 + (lane & 15);

    v8f acc = {0.f, 0.f, 0.f, 0.f, 0.f, 0.f, 0.f, 0.f};

#pragma unroll
    for (int kk = 0; kk < 5; ++kk) {
        int kb = kk * 32;
        // A fragment (16x32 bf16): per ISA layout, lane holds
        //   components 0..7  = K [half*8 .. half*8+7]      (contiguous 16B)
        //   components 8..15 = K [16+half*8 .. 16+half*8+7] (contiguous 16B)
        const __bf16* ap = hcat + (size_t)m * DIN + kb + half * 8;
        v8bf alo = *(const v8bf*)(ap);
        v8bf ahi = *(const v8bf*)(ap + 16);
        v16bf a = __builtin_shufflevector(alo, ahi,
                    0, 1, 2, 3, 4, 5, 6, 7, 8, 9, 10, 11, 12, 13, 14, 15);
        // B fragment (32x16 bf16): lane = N column, components follow the same K order as A
        v16bf b;
#pragma unroll
        for (int s = 0; s < 16; ++s) {
            int krow = kb + (s < 8 ? half * 8 + s : 16 + half * 8 + (s - 8));
            b[s] = Wr[(size_t)krow * DOUT + tn * 16 + (lane & 15)];
        }

        acc = __builtin_amdgcn_wmma_f32_16x16x32_bf16(
                  false, a, false, b, (short)0, acc, false, false);
    }

    // C/D layout: VGPR v, lane L -> M = v + (L>=16 ? 8 : 0), N = L%16
    int nO = tn * 16 + (lane & 15);
    int mB = tm * 16 + half * 8;
#pragma unroll
    for (int v = 0; v < 8; ++v)
        z[(size_t)(mB + v) * DOUT + nO] = acc[v];
}

// el/er: per (node, head) dot of z row (32) with attn vectors
__global__ void k_el_er(const float* __restrict__ z,
                        const float* __restrict__ al,
                        const float* __restrict__ ar,
                        float* __restrict__ el, float* __restrict__ er) {
    int i = blockIdx.x * blockDim.x + threadIdx.x;
    if (i >= N_NODES * NH) return;
    int n = i >> 2, h = i & 3;
    const float* zr  = z  + (size_t)n * DOUT + h * HD;
    const float* alr = al + h * HD;
    const float* arr = ar + h * HD;
    float sl = 0.f, sr = 0.f;
#pragma unroll
    for (int d = 0; d < HD; ++d) { sl += zr[d] * alr[d]; sr += zr[d] * arr[d]; }
    el[i] = sl;
    er[i] = sr;
}

__device__ __forceinline__ float edge_logit(const float* el, const float* er,
                                            int sn, int dn, int h) {
    float x = el[sn * NH + h] + er[dn * NH + h];
    return x >= 0.f ? x : NEG_SLOPE * x;
}

__global__ void k_edge_max(const int* __restrict__ src, const int* __restrict__ dst,
                           const float* __restrict__ el, const float* __restrict__ er,
                           unsigned* __restrict__ menc) {
    int i = blockIdx.x * blockDim.x + threadIdx.x;
    if (i >= NEDGE * NH) return;
    int e = i >> 2, h = i & 3;
    int sn = src[e], dn = dst[e];
    float x = edge_logit(el, er, sn, dn, h);
    atomicMax(menc + dn * NH + h, fenc(x));
}

__global__ void k_edge_expsum(const int* __restrict__ src, const int* __restrict__ dst,
                              const float* __restrict__ el, const float* __restrict__ er,
                              const unsigned* __restrict__ menc,
                              float* __restrict__ sbuf) {
    int i = blockIdx.x * blockDim.x + threadIdx.x;
    if (i >= NEDGE * NH) return;
    int e = i >> 2, h = i & 3;
    int sn = src[e], dn = dst[e];
    float x = edge_logit(el, er, sn, dn, h);
    float m = fdec(menc[dn * NH + h]);
    if (!isfinite(m)) m = 0.f;
    atomicAdd(sbuf + dn * NH + h, expf(x - m));
}

// one wave per edge; lane handles component (h*32 + lane) for h = 0..3
__global__ void k_edge_scatter(const int* __restrict__ src, const int* __restrict__ dst,
                               const float* __restrict__ el, const float* __restrict__ er,
                               const unsigned* __restrict__ menc,
                               const float* __restrict__ sbuf,
                               const float* __restrict__ z,
                               float* __restrict__ acc) {
    int tid  = blockIdx.x * blockDim.x + threadIdx.x;
    int e    = tid >> 5;
    int lane = tid & 31;
    if (e >= NEDGE) return;
    int sn = src[e], dn = dst[e];
    float alpha[NH];
#pragma unroll
    for (int h = 0; h < NH; ++h) {
        float x = edge_logit(el, er, sn, dn, h);
        float m = fdec(menc[dn * NH + h]);
        if (!isfinite(m)) m = 0.f;
        float ex = expf(x - m);
        float ss = sbuf[dn * NH + h];
        alpha[h] = ex / (ss > 0.f ? ss : 1.f);
    }
#pragma unroll
    for (int h = 0; h < NH; ++h) {
        int c = h * HD + lane;
        atomicAdd(&acc[(size_t)dn * DOUT + c], z[(size_t)sn * DOUT + c] * alpha[h]);
    }
}

// mean over R + bias-mean + relu.  last==0: write bf16 activations for next
// layer (time-enc cols untouched).  last==1: write f32 final features.
__global__ void k_finalize(const float* __restrict__ acc,
                           const float* __restrict__ bias_l,
                           __bf16* __restrict__ hcat,
                           float*  __restrict__ hfin, int last) {
    int i = blockIdx.x * blockDim.x + threadIdx.x;
    if (i >= N_NODES * DOUT) return;
    int n = i >> 7, c = i & 127;
    float bm = 0.f;
#pragma unroll
    for (int r = 0; r < R; ++r) bm += bias_l[r * DOUT + c];
    float v = (acc[i] + bm) * (1.0f / R);
    v = v > 0.f ? v : 0.f;
    if (last) hfin[i] = v;
    else      hcat[(size_t)n * DIN + c] = (__bf16)v;
}

__global__ void k_gather_out(const float* __restrict__ hfin,
                             const int* __restrict__ root,
                             float* __restrict__ out) {
    int i = blockIdx.x * blockDim.x + threadIdx.x;
    if (i >= NROOT * DOUT) return;
    int j = i >> 7, c = i & 127;
    out[i] = hfin[(size_t)root[j] * DOUT + c];
}

// ------------------------------------------------------------------ launch --

extern "C" void kernel_launch(void* const* d_in, const int* in_sizes, int n_in,
                              void* d_out, int out_size, void* d_ws, size_t ws_size,
                              hipStream_t stream) {
    const float* emb   = (const float*)d_in[0];
    const float* freq  = (const float*)d_in[1];
    const float* phase = (const float*)d_in[2];
    const float* W     = (const float*)d_in[3];   // (2,R,160,128)
    const float* al    = (const float*)d_in[4];   // (2,R,4,32)
    const float* ar    = (const float*)d_in[5];   // (2,R,4,32)
    const float* bias  = (const float*)d_in[6];   // (2,R,128)
    const int*   nid   = (const int*)d_in[7];
    const int*   esrc  = (const int*)d_in[8];     // (R,100000)
    const int*   edst  = (const int*)d_in[9];     // (R,100000)
    const int*   root  = (const int*)d_in[10];
    const int*   tsp   = (const int*)d_in[11];
    float*       out   = (float*)d_out;

    // workspace carve (256B aligned)
    char* wsp = (char*)d_ws;
    size_t off = 0;
    auto carve = [&](size_t bytes) -> char* {
        char* p = wsp + off;
        off += (bytes + 255) & ~(size_t)255;
        return p;
    };
    __bf16*   hcat = (__bf16*)  carve((size_t)N_NODES * DIN * 2);
    __bf16*   Wbf  = (__bf16*)  carve((size_t)R * DIN * DOUT * 2);
    float*    zbuf = (float*)   carve((size_t)N_NODES * DOUT * 4);
    float*    elb  = (float*)   carve((size_t)N_NODES * NH * 4);
    float*    erb  = (float*)   carve((size_t)N_NODES * NH * 4);
    unsigned* menc = (unsigned*)carve((size_t)N_NODES * NH * 4);
    float*    sbuf = (float*)   carve((size_t)N_NODES * NH * 4);
    float*    accb = (float*)   carve((size_t)N_NODES * DOUT * 4);
    float*    hfin = (float*)   carve((size_t)N_NODES * DOUT * 4);

    const int B = 256;
    auto blocks = [](long long n, int b) { return (int)((n + b - 1) / b); };

    k_build_hcat<<<blocks((long long)N_NODES * DIN, B), B, 0, stream>>>(
        emb, nid, freq, phase, tsp, hcat);

    for (int l = 0; l < 2; ++l) {
        k_convert_w<<<blocks(R * DIN * DOUT, B), B, 0, stream>>>(
            W + (size_t)l * R * DIN * DOUT, Wbf, R * DIN * DOUT);
        k_zero_f32<<<blocks((long long)N_NODES * DOUT, B), B, 0, stream>>>(
            accb, N_NODES * DOUT);

        for (int r = 0; r < R; ++r) {
            k_init_ms<<<blocks(N_NODES * NH, B), B, 0, stream>>>(menc, sbuf);

            // 25000 16x16 tiles, 4 waves per block
            k_gemm_z<<<(N_NODES / 16) * (DOUT / 16) / 4, 128, 0, stream>>>(
                hcat, Wbf + (size_t)r * DIN * DOUT, zbuf);

            k_el_er<<<blocks(N_NODES * NH, B), B, 0, stream>>>(
                zbuf, al + ((size_t)l * R + r) * NH * HD,
                      ar + ((size_t)l * R + r) * NH * HD, elb, erb);

            const int* sr = esrc + (size_t)r * NEDGE;
            const int* dr = edst + (size_t)r * NEDGE;
            k_edge_max<<<blocks(NEDGE * NH, B), B, 0, stream>>>(sr, dr, elb, erb, menc);
            k_edge_expsum<<<blocks(NEDGE * NH, B), B, 0, stream>>>(sr, dr, elb, erb, menc, sbuf);
            k_edge_scatter<<<blocks((long long)NEDGE * 32, B), B, 0, stream>>>(
                sr, dr, elb, erb, menc, sbuf, zbuf, accb);
        }

        k_finalize<<<blocks((long long)N_NODES * DOUT, B), B, 0, stream>>>(
            accb, bias + (size_t)l * R * DOUT, hcat, hfin, l == 1 ? 1 : 0);
    }

    k_gather_out<<<blocks((long long)NROOT * DOUT, B), B, 0, stream>>>(hfin, root, out);
}
